// GraphMamba_47339129537010
// MI455X (gfx1250) — matrix operand order repaired and checked by reading, not compiled, not run on hardware
//
#include <hip/hip_runtime.h>
#include <hip/hip_bf16.h>

#define N_ENT   50000
#define N_EDGE  600000
#define DIM     128
#define TWO_D   256
#define NLAYER  2
#define CHUNKS  400
#define RPC     125     // rows per chunk: 400*125 = 50000
#define ROWT    3125    // 50000/16 row tiles

typedef __attribute__((ext_vector_type(16))) __bf16 v16bf;
typedef __attribute__((ext_vector_type(8)))  __bf16 v8bf;
typedef __attribute__((ext_vector_type(8)))  float  v8f;

__device__ __forceinline__ float sigmoidf(float x) { return 1.0f / (1.0f + __expf(-x)); }

// ---------------- Phase 1: edge gather + atomic scatter-add -----------------
__global__ void edge_agg(const int* __restrict__ eidx, const int* __restrict__ etype,
                         const int* __restrict__ etime, const float* __restrict__ ew,
                         const float* __restrict__ ent, const float* __restrict__ rel,
                         const float* __restrict__ tim, float* __restrict__ sums,
                         float* __restrict__ cnt) {
  int e = blockIdx.x * 8 + (threadIdx.x >> 5);
  int lane = threadIdx.x & 31;
  int src = eidx[e];
  int dst = eidx[N_EDGE + e];
  int rt = etype[e];
  int tt = etime[e];
  float w = ew[e];
  float4 a = ((const float4*)(ent + (size_t)src * DIM))[lane];
  float4 b = ((const float4*)(rel + (size_t)rt  * DIM))[lane];
  float4 c = ((const float4*)(tim + (size_t)tt  * DIM))[lane];
  float* out = sums + (size_t)dst * DIM + lane * 4;
  atomicAdd(out + 0, (a.x + b.x + c.x) * w);
  atomicAdd(out + 1, (a.y + b.y + c.y) * w);
  atomicAdd(out + 2, (a.z + b.z + c.z) * w);
  atomicAdd(out + 3, (a.w + b.w + c.w) * w);
  if (lane == 0) atomicAdd(cnt + dst, w);
}

// ---------------- Phase 2: scatter-mean, L2 normalize, permute --------------
__global__ void mean_norm_perm(const float* __restrict__ sums, const float* __restrict__ cnt,
                               const int* __restrict__ perm, float* __restrict__ x,
                               __bf16* __restrict__ xbf) {
  int row = blockIdx.x * 8 + (threadIdx.x >> 5);
  int lane = threadIdx.x & 31;
  int p = perm[row];
  float inv_c = 1.0f / fmaxf(cnt[p], 1.0f);
  float4 v = ((const float4*)(sums + (size_t)p * DIM))[lane];
  v.x *= inv_c; v.y *= inv_c; v.z *= inv_c; v.w *= inv_c;
  float ss = v.x * v.x + v.y * v.y + v.z * v.z + v.w * v.w;
  #pragma unroll
  for (int m = 16; m >= 1; m >>= 1) ss += __shfl_xor(ss, m, 32);
  float inv = 1.0f / fmaxf(sqrtf(ss), 1e-12f);
  v.x *= inv; v.y *= inv; v.z *= inv; v.w *= inv;
  ((float4*)(x + (size_t)row * DIM))[lane] = v;
  __bf16* ob = xbf + (size_t)row * DIM + lane * 4;
  ob[0] = (__bf16)v.x; ob[1] = (__bf16)v.y; ob[2] = (__bf16)v.z; ob[3] = (__bf16)v.w;
}

// ---------------- Weight transpose + bf16 conversion ------------------------
__global__ void conv_w(const float* __restrict__ W, __bf16* __restrict__ WT, int ncols) {
  // W: [L][DIM][ncols] -> WT: [L][ncols][DIM]
  int idx = blockIdx.x * 256 + threadIdx.x;
  int tot = NLAYER * DIM * ncols;
  if (idx >= tot) return;
  int l = idx / (DIM * ncols);
  int rem = idx - l * DIM * ncols;
  int o = rem / DIM;
  int i = rem - o * DIM;
  WT[idx] = (__bf16)W[(size_t)l * DIM * ncols + (size_t)i * ncols + o];
}

// ---------------- WMMA bf16 GEMM: out = A @ B^T(+bias)(+resid) --------------
// Abf: [Mrows][128] bf16 row-major, BT: [NCOLS][128] bf16 (pre-transposed).
// One wave per 16x16 C tile; K=128 => 4 x v_wmma_f32_16x16x32_bf16.
// NCOLS compile-time => store offsets become instruction immediates;
// HAS_RESID compile-time => no branchy epilogue.
template <int NCOLS, bool HAS_RESID>
__global__ void gemm_bf16(const __bf16* __restrict__ Abf, const __bf16* __restrict__ BT,
                          const float* __restrict__ bias, const float* __restrict__ resid,
                          float* __restrict__ out) {
  int wave  = threadIdx.x >> 5;
  int lane  = threadIdx.x & 31;
  int row0  = blockIdx.x * 16;
  int col0  = (blockIdx.y * 8 + wave) * 16;
  int mn    = lane & 15;        // A row within tile / C column within tile
  int khalf = lane >> 4;        // ISA 16-bit A layout: half-wave K split
  v8f acc = {};
  const __bf16* arow = Abf + (size_t)(row0 + mn) * DIM + khalf * 8;
  const __bf16* brow = BT  + (size_t)(col0 + mn) * DIM + khalf * 8;
  #pragma unroll
  for (int kk = 0; kk < DIM; kk += 32) {
    v8bf alo = *(const v8bf*)(arow + kk);
    v8bf ahi = *(const v8bf*)(arow + kk + 16);
    v8bf blo = *(const v8bf*)(brow + kk);
    v8bf bhi = *(const v8bf*)(brow + kk + 16);
    v16bf a, b;
    #pragma unroll
    for (int e = 0; e < 8; e++) {
      a[e] = alo[e]; a[e + 8] = ahi[e];
      b[e] = blo[e]; b[e + 8] = bhi[e];
    }
    acc = __builtin_amdgcn_wmma_f32_16x16x32_bf16(false, a, false, b,
                                                  (short)0, acc, false, false);
  }
  float bv = bias[col0 + mn];
  // ISA f32 C/D layout: VGPR v holds M = v + 8*khalf, N = lane&15.
  size_t base = (size_t)(row0 + 8 * khalf) * NCOLS + col0 + mn;
  float*       orow = out + base;
  const float* rrow = HAS_RESID ? (resid + base) : nullptr;
  #pragma unroll
  for (int v = 0; v < 8; v++) {
    float val = acc[v] + bv;
    if (HAS_RESID) val += rrow[v * NCOLS];
    orow[v * NCOLS] = val;
  }
}

// ---------------- 3-pass chunked linear-recurrence scan ---------------------
// h_i = (sigmoid(g_i)*A[d]) * h_{i-1} + B[d]*z_i   over 50000 rows, per dim.
__global__ void scan_pass1(const float* __restrict__ zg, const float* __restrict__ Av,
                           const float* __restrict__ Bv, float* __restrict__ P,
                           float* __restrict__ Q) {
  int d = threadIdx.x, c = blockIdx.x;
  float Ad = Av[d], Bd = Bv[d];
  float p = 1.0f, q = 0.0f;
  const float* base = zg + (size_t)c * RPC * TWO_D;
  for (int i = 0; i < RPC; i++) {
    if (((d & 31) == 0) && (i + 16 < RPC)) {
      __builtin_prefetch(base + (size_t)(i + 16) * TWO_D + d, 0, 1);
      __builtin_prefetch(base + (size_t)(i + 16) * TWO_D + DIM + d, 0, 1);
    }
    float zv = base[(size_t)i * TWO_D + d];
    float gv = base[(size_t)i * TWO_D + DIM + d];
    float a = sigmoidf(gv) * Ad;
    q = a * q + Bd * zv;
    p *= a;
  }
  P[c * DIM + d] = p;
  Q[c * DIM + d] = q;
}

__global__ void scan_pass2(const float* __restrict__ P, const float* __restrict__ Q,
                           float* __restrict__ H) {
  int d = threadIdx.x;
  float carry = 0.0f;
  for (int c = 0; c < CHUNKS; c++) {
    H[c * DIM + d] = carry;
    carry = P[c * DIM + d] * carry + Q[c * DIM + d];
  }
}

__global__ void scan_pass3(const float* __restrict__ zg, const float* __restrict__ Av,
                           const float* __restrict__ Bv, const float* __restrict__ H,
                           __bf16* __restrict__ hbf) {
  int d = threadIdx.x, c = blockIdx.x;
  float Ad = Av[d], Bd = Bv[d];
  float h = H[c * DIM + d];
  const float* base = zg + (size_t)c * RPC * TWO_D;
  __bf16* ob = hbf + (size_t)c * RPC * DIM;
  for (int i = 0; i < RPC; i++) {
    if (((d & 31) == 0) && (i + 16 < RPC)) {
      __builtin_prefetch(base + (size_t)(i + 16) * TWO_D + d, 0, 1);
      __builtin_prefetch(base + (size_t)(i + 16) * TWO_D + DIM + d, 0, 1);
    }
    float zv = base[(size_t)i * TWO_D + d];
    float gv = base[(size_t)i * TWO_D + DIM + d];
    h = sigmoidf(gv) * Ad * h + Bd * zv;
    ob[(size_t)i * DIM + d] = (__bf16)h;
  }
}

// ---------------- LayerNorm (wave32 per row) --------------------------------
__global__ void layernorm(const float* __restrict__ y, const float* __restrict__ g,
                          const float* __restrict__ b, float* __restrict__ xout,
                          __bf16* __restrict__ xbf) {
  int row = blockIdx.x * 8 + (threadIdx.x >> 5);
  int lane = threadIdx.x & 31;
  float4 v = ((const float4*)(y + (size_t)row * DIM))[lane];
  float s = v.x + v.y + v.z + v.w;
  #pragma unroll
  for (int m = 16; m >= 1; m >>= 1) s += __shfl_xor(s, m, 32);
  float mu = s * (1.0f / DIM);
  float dx = v.x - mu, dy = v.y - mu, dz = v.z - mu, dw = v.w - mu;
  float s2 = dx * dx + dy * dy + dz * dz + dw * dw;
  #pragma unroll
  for (int m = 16; m >= 1; m >>= 1) s2 += __shfl_xor(s2, m, 32);
  float inv = rsqrtf(s2 * (1.0f / DIM) + 1e-5f);
  float4 gg = ((const float4*)g)[lane];
  float4 bb = ((const float4*)b)[lane];
  float4 o;
  o.x = dx * inv * gg.x + bb.x;
  o.y = dy * inv * gg.y + bb.y;
  o.z = dz * inv * gg.z + bb.z;
  o.w = dw * inv * gg.w + bb.w;
  ((float4*)(xout + (size_t)row * DIM))[lane] = o;
  __bf16* ob = xbf + (size_t)row * DIM + lane * 4;
  ob[0] = (__bf16)o.x; ob[1] = (__bf16)o.y; ob[2] = (__bf16)o.z; ob[3] = (__bf16)o.w;
}

// ---------------- Host side -------------------------------------------------
extern "C" void kernel_launch(void* const* d_in, const int* in_sizes, int n_in,
                              void* d_out, int out_size, void* d_ws, size_t ws_size,
                              hipStream_t stream) {
  (void)in_sizes; (void)n_in; (void)out_size; (void)ws_size;
  const int*   edge_index = (const int*)d_in[0];
  const int*   edge_type  = (const int*)d_in[1];
  const int*   edge_time  = (const int*)d_in[2];
  const float* edge_w     = (const float*)d_in[3];
  const int*   perm       = (const int*)d_in[4];
  const float* ent        = (const float*)d_in[5];
  const float* rel        = (const float*)d_in[6];
  const float* tim        = (const float*)d_in[7];
  const float* Wi         = (const float*)d_in[8];
  const float* bi         = (const float*)d_in[9];
  const float* Wo         = (const float*)d_in[10];
  const float* bo         = (const float*)d_in[11];
  const float* Av         = (const float*)d_in[12];
  const float* Bv         = (const float*)d_in[13];
  const float* ln_g       = (const float*)d_in[14];
  const float* ln_b       = (const float*)d_in[15];

  char* p = (char*)d_ws;
  auto take = [&](size_t bytes) { char* r = p; p += (bytes + 255) & ~(size_t)255; return r; };
  float*  sums = (float*) take((size_t)N_ENT * DIM * 4);
  float*  cnt  = (float*) take((size_t)N_ENT * 4);
  float*  x    = (float*) take((size_t)N_ENT * DIM * 4);
  __bf16* xbf  = (__bf16*)take((size_t)N_ENT * DIM * 2);
  float*  zg   = (float*) take((size_t)N_ENT * TWO_D * 4);
  __bf16* hbf  = (__bf16*)take((size_t)N_ENT * DIM * 2);
  float*  ybuf = (float*) take((size_t)N_ENT * DIM * 4);
  float*  Pc   = (float*) take((size_t)CHUNKS * DIM * 4);
  float*  Qc   = (float*) take((size_t)CHUNKS * DIM * 4);
  float*  Hc   = (float*) take((size_t)CHUNKS * DIM * 4);
  __bf16* WiT  = (__bf16*)take((size_t)NLAYER * TWO_D * DIM * 2);
  __bf16* WoT  = (__bf16*)take((size_t)NLAYER * DIM * DIM * 2);

  hipMemsetAsync(sums, 0, (size_t)N_ENT * DIM * 4, stream);
  hipMemsetAsync(cnt,  0, (size_t)N_ENT * 4, stream);

  edge_agg<<<N_EDGE / 8, 256, 0, stream>>>(edge_index, edge_type, edge_time, edge_w,
                                           ent, rel, tim, sums, cnt);
  mean_norm_perm<<<N_ENT / 8, 256, 0, stream>>>(sums, cnt, perm, x, xbf);
  conv_w<<<(NLAYER * DIM * TWO_D + 255) / 256, 256, 0, stream>>>(Wi, WiT, TWO_D);
  conv_w<<<(NLAYER * DIM * DIM + 255) / 256, 256, 0, stream>>>(Wo, WoT, DIM);

  for (int l = 0; l < NLAYER; l++) {
    gemm_bf16<TWO_D, false><<<dim3(ROWT, 2), 256, 0, stream>>>(
        xbf, WiT + (size_t)l * TWO_D * DIM, bi + l * TWO_D, nullptr, zg);
    scan_pass1<<<CHUNKS, DIM, 0, stream>>>(zg, Av + l * DIM, Bv + l * DIM, Pc, Qc);
    scan_pass2<<<1, DIM, 0, stream>>>(Pc, Qc, Hc);
    scan_pass3<<<CHUNKS, DIM, 0, stream>>>(zg, Av + l * DIM, Bv + l * DIM, Hc, hbf);
    gemm_bf16<DIM, true><<<dim3(ROWT, 1), 256, 0, stream>>>(
        hbf, WoT + (size_t)l * DIM * DIM, bo + l * DIM, x, ybuf);
    layernorm<<<N_ENT / 8, 256, 0, stream>>>(ybuf, ln_g + l * DIM, ln_b + l * DIM,
                                             (l == NLAYER - 1) ? (float*)d_out : x, xbf);
  }
}
